// Attention_84739704750279
// MI455X (gfx1250) — matrix operand order repaired and checked by reading, not compiled, not run on hardware
//
#include <hip/hip_runtime.h>
#include <hip/hip_bf16.h>

typedef __attribute__((ext_vector_type(16))) _Float16 v16h;
typedef __attribute__((ext_vector_type(8)))  _Float16 v8h;
typedef __attribute__((ext_vector_type(8)))  float    v8f;

#define NUM_HEAD 16
#define HEAD_DIM 64
#define DIM      1024
#define INNER    1024
#define QKV3     3072
#define SEQ      2048
#define BATCH    2
#define ROWS     (BATCH * SEQ)   /* 4096 */

__device__ __forceinline__ v16h cat16(v8h lo, v8h hi) {
  return __builtin_shufflevector(lo, hi, 0,1,2,3,4,5,6,7,8,9,10,11,12,13,14,15);
}

// D(f32 16x16) = A(f16 16x32) * B(f16 32x16) + C
__device__ __forceinline__ v8f wmma_f16(v16h a, v16h b, v8f c) {
  return __builtin_amdgcn_wmma_f32_16x16x32_f16(false, a, false, b, (short)0, c,
                                                false, false);
}

// Async global->LDS 16-byte copy (per-lane), tracked by ASYNCcnt.
// LDS byte address = low 32 bits of the generic pointer (LDS aperture rule).
__device__ __forceinline__ void cp_async_b128(const _Float16* gsrc, _Float16* ldst) {
  unsigned lds = (unsigned)(unsigned long long)ldst;
  unsigned long long ga = (unsigned long long)gsrc;
  asm volatile("global_load_async_to_lds_b128 %0, %1, off"
               :: "v"(lds), "v"(ga) : "memory");
}
__device__ __forceinline__ void wait_async0() {
  asm volatile("s_wait_asynccnt 0x0" ::: "memory");
}

// ------------------------------------------------- transpose + f32->f16
// dst[n][k] = (f16) src[k][n];  src: [K][N] f32. 32x32 LDS tiles.
__global__ __launch_bounds__(256)
void transpose_cvt(const float* __restrict__ src, _Float16* __restrict__ dst,
                   int K, int N) {
  __shared__ float tile[32][33];
  const int tx = threadIdx.x & 31, ty4 = (threadIdx.x >> 5) * 4;
  const int kt = blockIdx.y * 32, nt = blockIdx.x * 32;
#pragma unroll
  for (int i = 0; i < 4; ++i)
    tile[ty4 + i][tx] = src[(size_t)(kt + ty4 + i) * N + nt + tx];
  __syncthreads();
#pragma unroll
  for (int i = 0; i < 4; ++i)
    dst[(size_t)(nt + ty4 + i) * K + kt + tx] = (_Float16)tile[tx][ty4 + i];
}

// ---------------------------------------------------------------- layernorm
__global__ __launch_bounds__(256)
void ln_f16(const float* __restrict__ x, const float* __restrict__ g,
            const float* __restrict__ bb, _Float16* __restrict__ h) {
  __shared__ float ssum[8], ssum2[8];
  const int tid = threadIdx.x;
  const int lane = tid & 31, w = tid >> 5;
  const size_t row = blockIdx.x;
  const float* xr = x + row * DIM;
  float4 xv = ((const float4*)xr)[tid];
  float s  = xv.x + xv.y + xv.z + xv.w;
  float s2 = xv.x * xv.x + xv.y * xv.y + xv.z * xv.z + xv.w * xv.w;
#pragma unroll
  for (int off = 16; off > 0; off >>= 1) {
    s  += __shfl_xor(s,  off);
    s2 += __shfl_xor(s2, off);
  }
  if (lane == 0) { ssum[w] = s; ssum2[w] = s2; }
  __syncthreads();
  float S = 0.f, S2 = 0.f;
#pragma unroll
  for (int i = 0; i < 8; ++i) { S += ssum[i]; S2 += ssum2[i]; }
  const float mu  = S * (1.0f / DIM);
  const float var = S2 * (1.0f / DIM) - mu * mu;
  const float rs  = rsqrtf(var + 1e-5f);
  float4 gv = ((const float4*)g)[tid];
  float4 bv = ((const float4*)bb)[tid];
  _Float16* hr = h + row * DIM + tid * 4;
  hr[0] = (_Float16)((xv.x - mu) * rs * gv.x + bv.x);
  hr[1] = (_Float16)((xv.y - mu) * rs * gv.y + bv.y);
  hr[2] = (_Float16)((xv.z - mu) * rs * gv.z + bv.z);
  hr[3] = (_Float16)((xv.w - mu) * rs * gv.w + bv.w);
}

// ---------------------------------------------------------------- GEMM (TN)
// D = X[M,K] @ W, with W pre-transposed: Wt[N][K] (K-major).
// Computes D^T tiles: WMMA A-operand = Wt rows (M-dim=n), B-operand = X rows
// (N-dim=m). Both LDS tiles staged with straight async b128 copies.
// Block tile 128n x 128m, 8 waves = 4(n) x 2(m), wave 32n x 64m = 8 wmma.
// Double-buffered LDS, K-step 32. Per lane, the 8 acc values are contiguous
// in n -> vectorized stores. If vT != null and n0 >= 2048 (V third of QKV),
// scatter into vT[n-2048][m] (pre-transposed V for the attention kernel).
template <typename OutT>
__global__ __launch_bounds__(256)
void gemm_tn_wmma(const _Float16* __restrict__ Wt, const _Float16* __restrict__ X,
                  OutT* __restrict__ Dout, _Float16* __restrict__ vT,
                  int M, int N, int K) {
  __shared__ _Float16 sW[2][128 * 32];   // [n][k]
  __shared__ _Float16 sX[2][128 * 32];   // [m][k]
  const int tid = threadIdx.x, lane = tid & 31, w = tid >> 5;
  const int wn = w & 3, wm = w >> 2;
  const int hl = lane >> 4, l16 = lane & 15;
  const int n0 = blockIdx.y * 128, m0 = blockIdx.x * 128;
  v8f acc[2][4] = {};
  const int crow = tid >> 1, cc = (tid & 1) << 4;   // 128 rows, 2 x 16 halves
  const _Float16* Wg = Wt + (size_t)(n0 + crow) * K + cc;
  const _Float16* Xg = X  + (size_t)(m0 + crow) * K + cc;

  cp_async_b128(Wg,     &sW[0][crow * 32 + cc]);
  cp_async_b128(Wg + 8, &sW[0][crow * 32 + cc + 8]);
  cp_async_b128(Xg,     &sX[0][crow * 32 + cc]);
  cp_async_b128(Xg + 8, &sX[0][crow * 32 + cc + 8]);
  wait_async0();
  __syncthreads();

  int cb = 0;
  for (int k0 = 0; k0 < K; k0 += 32) {
    const int nb = cb ^ 1;
    if (k0 + 32 < K) {
      cp_async_b128(Wg + k0 + 32, &sW[nb][crow * 32 + cc]);
      cp_async_b128(Wg + k0 + 40, &sW[nb][crow * 32 + cc + 8]);
      cp_async_b128(Xg + k0 + 32, &sX[nb][crow * 32 + cc]);
      cp_async_b128(Xg + k0 + 40, &sX[nb][crow * 32 + cc + 8]);
    }
    // A-operand (Wt): lane = n row; elems 0-7: k=hl*8+i, 8-15: k=16+hl*8+i
    v16h wf[2];
#pragma unroll
    for (int s = 0; s < 2; ++s) {
      const _Float16* p = &sW[cb][(wn * 32 + s * 16 + l16) * 32 + hl * 8];
      wf[s] = cat16(*(const v8h*)p, *(const v8h*)(p + 16));
    }
#pragma unroll
    for (int t = 0; t < 4; ++t) {
      // B-operand (X): lane = m col; elems 0-7: k=hl*8+i, 8-15: k=16+hl*8+i
      const _Float16* p = &sX[cb][(wm * 64 + t * 16 + l16) * 32 + hl * 8];
      v16h xf = cat16(*(const v8h*)p, *(const v8h*)(p + 16));
#pragma unroll
      for (int s = 0; s < 2; ++s)
        acc[s][t] = wmma_f16(wf[s], xf, acc[s][t]);
    }
    wait_async0();
    __syncthreads();
    cb = nb;
  }

  const bool vpath = (vT != nullptr) && (n0 >= 2 * INNER);
#pragma unroll
  for (int s = 0; s < 2; ++s) {
    const int nbase = n0 + wn * 32 + s * 16 + hl * 8;
#pragma unroll
    for (int t = 0; t < 4; ++t) {
      const int m = m0 + wm * 64 + t * 16 + l16;
      if (vpath) {
#pragma unroll
        for (int r = 0; r < 8; ++r)
          vT[(size_t)(nbase + r - 2 * INNER) * M + m] = (_Float16)acc[s][t][r];
      } else if constexpr (sizeof(OutT) == 2) {
        v8h pk;
#pragma unroll
        for (int r = 0; r < 8; ++r) pk[r] = (_Float16)acc[s][t][r];
        *(v8h*)((_Float16*)Dout + (size_t)m * N + nbase) = pk;
      } else {
        float4 lo, hi;
        lo.x = acc[s][t][0]; lo.y = acc[s][t][1]; lo.z = acc[s][t][2]; lo.w = acc[s][t][3];
        hi.x = acc[s][t][4]; hi.y = acc[s][t][5]; hi.z = acc[s][t][6]; hi.w = acc[s][t][7];
        float* dp = (float*)Dout + (size_t)m * N + nbase;
        *(float4*)dp = lo;
        *(float4*)(dp + 4) = hi;
      }
    }
  }
}

// ---------------------------------------------------------------- attention
// qkv: [ROWS][3072] f16 (q|k row-major; v third unused here).
// vT:  [1024][ROWS] f16 (pre-transposed V: [h*64+d][token]).
// Grid: x = query block (128 q / block), y = b*16 + head.
// Each of 8 waves owns 16 query rows; 64-key chunks (16 wmma per chunk),
// double-buffered K/V staged via async b128. AV computed as O^T = V^T * P^T.
__global__ __launch_bounds__(256)
void attn_wmma(const _Float16* __restrict__ qkv, const _Float16* __restrict__ vT,
               _Float16* __restrict__ aout) {
  __shared__ _Float16 sK[2][64 * 64];     // [key][d]
  __shared__ _Float16 sVt[2][64 * 72];    // [d][key], padded stride 72
  __shared__ _Float16 sP[8 * 16 * 64];    // per-wave P tile [q][key]
  __shared__ float    sStat[8 * 16];      // per-wave per-query stats
  const int tid = threadIdx.x, lane = tid & 31, w = tid >> 5;
  const int hl = lane >> 4, l16 = lane & 15;
  const int b = blockIdx.y >> 4, h = blockIdx.y & 15;
  const int q0 = blockIdx.x * 128 + w * 16;
  const size_t base = (size_t)b * SEQ * QKV3 + (size_t)h * HEAD_DIM;

  // Preload Q fragments for d=[0,32) and d=[32,64); fold in 1/sqrt(64)
  v16h qf[2];
  {
    const _Float16* qp = qkv + base + (size_t)(q0 + l16) * QKV3;
#pragma unroll
    for (int g = 0; g < 2; ++g) {
      qf[g] = cat16(*(const v8h*)(qp + g * 32 + hl * 8),
                    *(const v8h*)(qp + g * 32 + 16 + hl * 8));
      qf[g] = qf[g] * (_Float16)0.125f;
    }
  }
  float mrow[8], lrow[8];
  v8f o[4] = {{}, {}, {}, {}};   // O^T tiles: lane col = q, rows = d
#pragma unroll
  for (int r = 0; r < 8; ++r) { mrow[r] = -3.0e38f; lrow[r] = 0.f; }

  const int krow = tid >> 2, kc16 = (tid & 3) << 4;  // K: 64 keys x 64 d
  const int vrow = tid >> 2, vc16 = (tid & 3) << 4;  // V^T: 64 d x 64 keys
  const _Float16* Kg = qkv + base + (size_t)krow * QKV3 + INNER + kc16;
  const _Float16* Vg = vT + (size_t)(h * HEAD_DIM + vrow) * ROWS + b * SEQ + vc16;

  cp_async_b128(Kg,     &sK[0][krow * 64 + kc16]);
  cp_async_b128(Kg + 8, &sK[0][krow * 64 + kc16 + 8]);
  cp_async_b128(Vg,     &sVt[0][vrow * 72 + vc16]);
  cp_async_b128(Vg + 8, &sVt[0][vrow * 72 + vc16 + 8]);
  wait_async0();
  __syncthreads();

  int cb = 0;
  for (int kc = 0; kc < SEQ; kc += 64) {
    const int nb = cb ^ 1;
    if (kc + 64 < SEQ) {
      cp_async_b128(Kg + (size_t)(kc + 64) * QKV3,     &sK[nb][krow * 64 + kc16]);
      cp_async_b128(Kg + (size_t)(kc + 64) * QKV3 + 8, &sK[nb][krow * 64 + kc16 + 8]);
      cp_async_b128(Vg + kc + 64,     &sVt[nb][vrow * 72 + vc16]);
      cp_async_b128(Vg + kc + 72,     &sVt[nb][vrow * 72 + vc16 + 8]);
    }

    // S = Q * K^T  (four 16x16 tiles over 64 keys; contraction d = 32+32)
    v8f s[4];
#pragma unroll
    for (int jt = 0; jt < 4; ++jt) {
      v8f a = {};
      const _Float16* kb = &sK[cb][(jt * 16 + l16) * 64];  // lane = key column
      a = wmma_f16(qf[0], cat16(*(const v8h*)(kb + hl * 8),
                                *(const v8h*)(kb + 16 + hl * 8)), a);
      a = wmma_f16(qf[1], cat16(*(const v8h*)(kb + 32 + hl * 8),
                                *(const v8h*)(kb + 48 + hl * 8)), a);
      s[jt] = a;
    }

    // online softmax (rows live across the 16-lane half-wave)
    float corr[8];
#pragma unroll
    for (int r = 0; r < 8; ++r) {
      float mx = fmaxf(fmaxf(s[0][r], s[1][r]), fmaxf(s[2][r], s[3][r]));
#pragma unroll
      for (int off = 1; off < 16; off <<= 1) mx = fmaxf(mx, __shfl_xor(mx, off, 16));
      float mn = fmaxf(mrow[r], mx);
      corr[r] = __expf(mrow[r] - mn);
      float p0 = __expf(s[0][r] - mn), p1 = __expf(s[1][r] - mn);
      float p2 = __expf(s[2][r] - mn), p3 = __expf(s[3][r] - mn);
      s[0][r] = p0; s[1][r] = p1; s[2][r] = p2; s[3][r] = p3;
      float rs = (p0 + p1) + (p2 + p3);
#pragma unroll
      for (int off = 1; off < 16; off <<= 1) rs += __shfl_xor(rs, off, 16);
      lrow[r] = lrow[r] * corr[r] + rs;
      mrow[r] = mn;
    }

    // exchange per-query corr: row (hl*8+r) stats -> lane column q=l16
#pragma unroll
    for (int r = 0; r < 8; ++r)
      if (l16 == r) sStat[w * 16 + hl * 8 + r] = corr[r];
    // P tile [q][key] (contiguous in key for the P^T B-operand read)
    _Float16* pw = sP + w * 1024;
#pragma unroll
    for (int jt = 0; jt < 4; ++jt)
#pragma unroll
      for (int r = 0; r < 8; ++r)
        pw[(hl * 8 + r) * 64 + jt * 16 + l16] = (_Float16)s[jt][r];

    float corrq = sStat[w * 16 + l16];   // in-order DS, same wave
#pragma unroll
    for (int t = 0; t < 4; ++t) o[t] = o[t] * corrq;

    // P^T B-operand: lane = q col; elems 0-7: key = g*32 + hl*8 + i, 8-15: key = g*32 + 16 + hl*8 + i
    v16h pf[2];
#pragma unroll
    for (int g = 0; g < 2; ++g) {
      const _Float16* pr = pw + l16 * 64 + g * 32 + hl * 8;
      pf[g] = cat16(*(const v8h*)pr, *(const v8h*)(pr + 16));
    }

    // O^T += V^T * P^T  (A-operand = V^T rows: lane = d, grouped-k pattern)
#pragma unroll
    for (int t = 0; t < 4; ++t) {
      const _Float16* vb = &sVt[cb][(t * 16 + l16) * 72 + hl * 8];
      v8f a = o[t];
      a = wmma_f16(cat16(*(const v8h*)vb, *(const v8h*)(vb + 16)), pf[0], a);
      a = wmma_f16(cat16(*(const v8h*)(vb + 32), *(const v8h*)(vb + 48)), pf[1], a);
      o[t] = a;
    }

    wait_async0();
    __syncthreads();
    cb = nb;
  }

#pragma unroll
  for (int r = 0; r < 8; ++r)
    if (l16 == r) sStat[w * 16 + hl * 8 + r] = lrow[r];
  const float linv = 1.0f / sStat[w * 16 + l16];

  // lane holds q = q0 + l16; per tile t the 8 d-values are contiguous
  _Float16* op = aout + ((size_t)b * SEQ + q0 + l16) * INNER + h * HEAD_DIM;
#pragma unroll
  for (int t = 0; t < 4; ++t) {
    v8h pk;
#pragma unroll
    for (int r = 0; r < 8; ++r) pk[r] = (_Float16)(o[t][r] * linv);
    *(v8h*)(op + t * 16 + hl * 8) = pk;
  }
}

// ---------------------------------------------------------------- launch
extern "C" void kernel_launch(void* const* d_in, const int* in_sizes, int n_in,
                              void* d_out, int out_size, void* d_ws, size_t ws_size,
                              hipStream_t stream) {
  const float* x     = (const float*)d_in[0];
  const float* gamma = (const float*)d_in[1];
  const float* beta  = (const float*)d_in[2];
  const float* wqkv  = (const float*)d_in[3];
  const float* wout  = (const float*)d_in[4];
  float* out = (float*)d_out;

  // workspace layout (f16): ~56 MB total
  _Float16* ws = (_Float16*)d_ws;
  _Float16* h16    = ws;  ws += (size_t)ROWS * DIM;     // x after LN, [m][k]
  _Float16* wqkvT  = ws;  ws += (size_t)QKV3 * DIM;     // [n][k]
  _Float16* woutT  = ws;  ws += (size_t)DIM * INNER;    // [n][k]
  _Float16* qkv16  = ws;  ws += (size_t)ROWS * QKV3;    // q|k row-major
  _Float16* vT     = ws;  ws += (size_t)INNER * ROWS;   // [h*64+d][token]
  _Float16* attn16 = ws;                                // [token][1024]

  transpose_cvt<<<dim3(QKV3 / 32, DIM / 32), 256, 0, stream>>>(wqkv, wqkvT, DIM, QKV3);
  transpose_cvt<<<dim3(DIM / 32, INNER / 32), 256, 0, stream>>>(wout, woutT, INNER, DIM);

  ln_f16<<<ROWS, 256, 0, stream>>>(x, gamma, beta, h16);

  // qkv = h @ w_qkv : q,k thirds -> qkv16 row-major; v third -> vT transposed
  gemm_tn_wmma<_Float16><<<dim3(ROWS / 128, QKV3 / 128), 256, 0, stream>>>(
      wqkvT, h16, qkv16, vT, ROWS, QKV3, DIM);

  attn_wmma<<<dim3(SEQ / 128, BATCH * NUM_HEAD), 256, 0, stream>>>(qkv16, vT, attn16);

  // out = attn @ w_out -> f32
  gemm_tn_wmma<float><<<dim3(ROWS / 128, DIM / 128), 256, 0, stream>>>(
      woutT, attn16, out, (_Float16*)nullptr, ROWS, DIM, INNER);
}
